// AttentionBlock_28656021799490
// MI455X (gfx1250) — matrix operand order repaired and checked by reading, not compiled, not run on hardware
//
#include <hip/hip_runtime.h>

// ---------------------------------------------------------------------------
// AttentionBlock for MI455X (gfx1250): bf16 WMMA matmuls, f32 softmax/epilogues
// GEMM tiles staged by the Tensor Data Mover (async DMA, TENSORcnt-tracked,
// double buffered). B=8, S=1024, D=1024, H=16, HD=64
// ---------------------------------------------------------------------------

typedef __bf16 bf16_t;
typedef __attribute__((ext_vector_type(16))) __bf16 v16bf;
typedef __attribute__((ext_vector_type(8)))  float  v8f;
typedef __attribute__((ext_vector_type(4)))  unsigned int v4u;
typedef __attribute__((ext_vector_type(8)))  int v8i;
typedef __attribute__((ext_vector_type(4)))  int v4i;

#if __has_builtin(__builtin_amdgcn_tensor_load_to_lds)
#define USE_TDM 1
#else
#define USE_TDM 0
#endif

// POD 16-byte chunk (avoid HIP vector-type ctors inside unions)
struct alignas(16) U4 { unsigned int x, y, z, w; };
union FragCvt { U4 u[2]; v16bf v; };

static __device__ __forceinline__ bf16_t f2bf(float f) {
  unsigned u = __float_as_uint(f);
  unsigned r = (u + 0x7FFFu + ((u >> 16) & 1u)) >> 16;   // round-to-nearest-even
  return __builtin_bit_cast(__bf16, (unsigned short)r);
}

// --- WMMA fragment loaders from LDS, per CDNA5 ISA 7.12.2 (wave32) -----------
// A (16x32 bf16): lane&15 = M; lanes 0-15 chunks K[0..7],K[16..23];
//                 lanes 16-31 chunks K[8..15],K[24..31]
static __device__ __forceinline__ v16bf load_fragA(const bf16_t* tile, int ld) {
  int lane = threadIdx.x & 31;
  int m  = lane & 15;
  int kb = (lane >> 4) << 3;            // 0 or 8
  const bf16_t* p = tile + m * ld + kb;
  FragCvt f;
  f.u[0] = *(const U4*)(p);
  f.u[1] = *(const U4*)(p + 16);
  return f.v;
}
// B (32x16 bf16): lane&15 = N; lanes 0-15 chunks K[0..7],K[16..23]; lanes 16-31 K[8..15],K[24..31]
static __device__ __forceinline__ v16bf load_fragB(const bf16_t* tile, int ld) {
  int lane = threadIdx.x & 31;
  int n  = lane & 15;
  int kb = (lane >> 4) << 3;            // 0 or 8
  const bf16_t* p = tile + n * ld + kb;
  FragCvt f;
  f.u[0] = *(const U4*)(p);
  f.u[1] = *(const U4*)(p + 16);
  return f.v;
}

static __device__ __forceinline__ v8f wmma_bf16(v16bf a, v16bf b, v8f c) {
  return __builtin_amdgcn_wmma_f32_16x16x32_bf16(false, a, false, b,
                                                 (short)0, c, false, false);
}

// ---------------------------------------------------------------------------
// TDM: async DMA of one 128-row x 32-col bf16 tile (row-major, row stride =
// K elements) from global into LDS, padded to a 40-element LDS row stride.
// Descriptor per CDNA5 ISA 8.3/8.4: 2D tensor, data_size=2B,
// pad_interval=16 DWORDs (one 32-elem row), pad_amount=4 DWORDs (8 elems).
// ---------------------------------------------------------------------------
#if USE_TDM
static __device__ __forceinline__ void
tdm_load_tile_128x32(const void* gsrc, unsigned lds_byte, unsigned k_elems,
                     unsigned rows_total) {
  unsigned long long ga = (unsigned long long)(uintptr_t)gsrc;
  // group0: count=1 | lds_addr | global_addr[56:0] | type=2 ("image")
  v4u g0 = { 1u,
             lds_byte,
             (unsigned)ga,
             (unsigned)((ga >> 32) & 0x01FFFFFFu) | 0x80000000u };
  unsigned dim0 = k_elems;       // tensor length along K (elements)
  unsigned dim1 = rows_total;    // tensor rows
  v8i g1 = {
    (int)((1u << 16)            // data_size = 2 bytes
          | (1u << 20)          // pad_enable
          | (3u << 22)          // pad_interval: 16 DWORDs
          | (3u << 25)),        // pad_amount:   4 DWORDs
    (int)((dim0 & 0xFFFFu) << 16),                           // dim0[15:0]
    (int)(((dim0 >> 16) & 0xFFFFu) | ((dim1 & 0xFFFFu) << 16)),
    (int)(((dim1 >> 16) & 0xFFFFu) | (32u << 16)),           // tile_dim0 = 32
    (int)128,                                                // tile_dim1 = 128
    (int)k_elems,                                            // dim0_stride
    0, 0 };
  v4i gz = { 0, 0, 0, 0 };
#if __clang_major__ >= 23
  v8i gz8 = { 0, 0, 0, 0, 0, 0, 0, 0 };
  __builtin_amdgcn_tensor_load_to_lds(g0, g1, gz, gz, gz8, 0);
#else
  __builtin_amdgcn_tensor_load_to_lds(g0, g1, gz, gz, 0);
#endif
}
#endif

// ---------------------------------------------------------------------------
// f32 -> bf16 conversion (bandwidth bound, 4-wide)
// ---------------------------------------------------------------------------
__global__ void __launch_bounds__(256)
cvt_f32_bf16(const float* __restrict__ s, bf16_t* __restrict__ d, long long n) {
  long long i = ((long long)blockIdx.x * blockDim.x + threadIdx.x) * 4;
  if (i + 3 >= n) return;
  float4 v = *(const float4*)(s + i);
  union { bf16_t e[4]; unsigned long long q; } o;
  o.e[0] = f2bf(v.x); o.e[1] = f2bf(v.y); o.e[2] = f2bf(v.z); o.e[3] = f2bf(v.w);
  *(unsigned long long*)(d + i) = o.q;
}

// ---------------------------------------------------------------------------
// Shared GEMM core: C[128x128] = A[128xK] * B[128xK]^T  (both K-major bf16)
// 256 threads = 8 waves (2x4), each wave owns 64x32 (4x2 WMMA accumulators).
// K staged 32 at a time into double-buffered LDS by the TDM; wave 0 issues
// next-slab DMAs while all waves compute the current slab.
// ---------------------------------------------------------------------------
#define G_LD  40                        // 32 + 8 pad elems; 80B rows (16B mult)
#define G_BUF (128 * G_LD)              // elements per tile buffer

static __device__ __forceinline__ void
gemm_core(const bf16_t* __restrict__ A, const bf16_t* __restrict__ Bw,
          int K, int rowsA, int rowsB, size_t m0, size_t n0,
          bf16_t* As, bf16_t* Bs, v8f acc[4][2]) {
  const int tid  = threadIdx.x;
  const int warp = tid >> 5;
  const int wm   = (warp >> 2) * 64;    // 0 / 64
  const int wn   = (warp & 3) * 32;     // 0 / 32 / 64 / 96

  for (int mi = 0; mi < 4; ++mi)
    for (int nj = 0; nj < 2; ++nj)
      for (int r = 0; r < 8; ++r) acc[mi][nj][r] = 0.0f;

#if USE_TDM
  const unsigned asBase = (unsigned)(uintptr_t)As;
  const unsigned bsBase = (unsigned)(uintptr_t)Bs;
  const unsigned bufBytes = G_BUF * 2;

  if (warp == 0) {    // prime buffer 0 (tensor ops ignore EXEC; one wave issues)
    tdm_load_tile_128x32(A  + m0 * (size_t)K, asBase, (unsigned)K, rowsA);
    tdm_load_tile_128x32(Bw + n0 * (size_t)K, bsBase, (unsigned)K, rowsB);
  }
  int cur = 0;
  for (int k0 = 0; k0 < K; k0 += 32, cur ^= 1) {
    if (warp == 0) {
      if (k0 + 32 < K) {
        unsigned nb = (unsigned)(cur ^ 1);
        tdm_load_tile_128x32(A  + m0 * (size_t)K + k0 + 32,
                             asBase + nb * bufBytes, (unsigned)K, rowsA);
        tdm_load_tile_128x32(Bw + n0 * (size_t)K + k0 + 32,
                             bsBase + nb * bufBytes, (unsigned)K, rowsB);
        __builtin_amdgcn_s_wait_tensorcnt(2);   // retire current buffer's pair
      } else {
        __builtin_amdgcn_s_wait_tensorcnt(0);
      }
    }
    __syncthreads();                            // data visible to all waves
    const bf16_t* Ab = &As[cur * G_BUF];
    const bf16_t* Bb = &Bs[cur * G_BUF];
    v16bf af[4], bfr[2];
    #pragma unroll
    for (int mi = 0; mi < 4; ++mi)
      af[mi] = load_fragA(&Ab[(wm + mi * 16) * G_LD], G_LD);
    #pragma unroll
    for (int nj = 0; nj < 2; ++nj)
      bfr[nj] = load_fragB(&Bb[(wn + nj * 16) * G_LD], G_LD);
    #pragma unroll
    for (int mi = 0; mi < 4; ++mi)
      #pragma unroll
      for (int nj = 0; nj < 2; ++nj)
        acc[mi][nj] = wmma_bf16(af[mi], bfr[nj], acc[mi][nj]);
    __syncthreads();           // all reads of `cur` done before TDM reuses it
  }
#else
  // Fallback: manual single-buffered staging
  for (int k0 = 0; k0 < K; k0 += 32) {
    __syncthreads();
    #pragma unroll
    for (int i = 0; i < 2; ++i) {
      int idx = (tid + i * 256) * 8;
      int row = idx >> 5, col = idx & 31;
      *(U4*)&As[row * G_LD + col] = *(const U4*)&A[(m0 + row) * (size_t)K + k0 + col];
      *(U4*)&Bs[row * G_LD + col] = *(const U4*)&Bw[(n0 + row) * (size_t)K + k0 + col];
    }
    __syncthreads();
    v16bf af[4], bfr[2];
    #pragma unroll
    for (int mi = 0; mi < 4; ++mi)
      af[mi] = load_fragA(&As[(wm + mi * 16) * G_LD], G_LD);
    #pragma unroll
    for (int nj = 0; nj < 2; ++nj)
      bfr[nj] = load_fragB(&Bs[(wn + nj * 16) * G_LD], G_LD);
    #pragma unroll
    for (int mi = 0; mi < 4; ++mi)
      #pragma unroll
      for (int nj = 0; nj < 2; ++nj)
        acc[mi][nj] = wmma_bf16(af[mi], bfr[nj], acc[mi][nj]);
  }
#endif
}

// ---------------------------------------------------------------------------
// Kernel: QKV projection. qkv = x @ Wi^T + bi, scattered as bf16 [B,H,S,HD]
// ---------------------------------------------------------------------------
__global__ void __launch_bounds__(256)
gemm_qkv(const bf16_t* __restrict__ xb, const bf16_t* __restrict__ Wib,
         const float* __restrict__ bi,
         bf16_t* __restrict__ Qb, bf16_t* __restrict__ Kb, bf16_t* __restrict__ Vb) {
  __shared__ __align__(16) bf16_t As[2 * G_BUF];
  __shared__ __align__(16) bf16_t Bs[2 * G_BUF];
  const size_t n0 = (size_t)blockIdx.x * 128;
  const size_t m0 = (size_t)blockIdx.y * 128;
  v8f acc[4][2];
  gemm_core(xb, Wib, 1024, 8192, 3072, m0, n0, As, Bs, acc);

  const int lane = threadIdx.x & 31;
  const int warp = threadIdx.x >> 5;
  const int wm = (warp >> 2) * 64, wn = (warp & 3) * 32;
  const int halfsel = (lane >> 4) << 3;
  const int col16   = lane & 15;
  #pragma unroll
  for (int mi = 0; mi < 4; ++mi)
    #pragma unroll
    for (int nj = 0; nj < 2; ++nj)
      #pragma unroll
      for (int r = 0; r < 8; ++r) {
        int gm = (int)m0 + wm + mi * 16 + r + halfsel;      // row in [0, B*S)
        int gn = (int)n0 + wn + nj * 16 + col16;            // col in [0, 3D)
        float v = acc[mi][nj][r] + bi[gn];
        int comp = gn >> 10;                                // 0=q 1=k 2=v
        int hc = gn & 1023, hh = hc >> 6, d = hc & 63;
        int bb = gm >> 10, s = gm & 1023;
        size_t o = (((size_t)(bb * 16 + hh)) * 1024 + s) * 64 + d;
        bf16_t* dst = (comp == 0) ? Qb : (comp == 1) ? Kb : Vb;
        dst[o] = f2bf(v);
      }
}

// ---------------------------------------------------------------------------
// Kernel: fused flash attention per (b, h, 64-row q tile). 4 waves, each wave
// owns 16 q-rows. Online softmax with additive log(clamp(cutoff)) mask.
// ---------------------------------------------------------------------------
#define ATT_LD 72
__global__ void __launch_bounds__(128)
attn_kernel(const bf16_t* __restrict__ Qb, const bf16_t* __restrict__ Kb,
            const bf16_t* __restrict__ Vb, const float* __restrict__ cutoff,
            bf16_t* __restrict__ attnb) {
  __shared__ __align__(16) bf16_t Qs [64 * ATT_LD];
  __shared__ __align__(16) bf16_t Ks [64 * ATT_LD];
  __shared__ __align__(16) bf16_t Vts[64 * ATT_LD];   // V transposed: [d][k]
  __shared__ __align__(16) bf16_t Ps [64 * ATT_LD];   // probabilities (bf16)

  const int tid  = threadIdx.x;
  const int lane = tid & 31;
  const int warp = tid >> 5;
  const int h = blockIdx.y, b = blockIdx.z;
  const int q0 = blockIdx.x * 64;
  const size_t headBase = ((size_t)(b * 16 + h)) * 1024 * 64;   // [B,H,S,HD]

  // stage Q tile (64x64)
  #pragma unroll
  for (int i = 0; i < 4; ++i) {
    int idx = (tid + i * 128) * 8;
    int row = idx >> 6, col = idx & 63;
    *(U4*)&Qs[row * ATT_LD + col] =
        *(const U4*)&Qb[headBase + (size_t)(q0 + row) * 64 + col];
  }

  v8f Of[4];
  #pragma unroll
  for (int n = 0; n < 4; ++n)
    for (int r = 0; r < 8; ++r) Of[n][r] = 0.0f;
  float mrow[8], lrow[8];
  #pragma unroll
  for (int r = 0; r < 8; ++r) { mrow[r] = -1e30f; lrow[r] = 0.0f; }

  const int qw = warp * 16;
  const int halfsel = (lane >> 4) << 3;
  const int col16   = lane & 15;

  for (int kv0 = 0; kv0 < 1024; kv0 += 64) {
    __syncthreads();
    // stage K tile row-major, V tile transposed
    #pragma unroll
    for (int i = 0; i < 4; ++i) {
      int idx = (tid + i * 128) * 8;
      int row = idx >> 6, col = idx & 63;
      *(U4*)&Ks[row * ATT_LD + col] =
          *(const U4*)&Kb[headBase + (size_t)(kv0 + row) * 64 + col];
      union { U4 u; bf16_t e[8]; } vv;
      vv.u = *(const U4*)&Vb[headBase + (size_t)(kv0 + row) * 64 + col];
      #pragma unroll
      for (int e = 0; e < 8; ++e)
        Vts[(col + e) * ATT_LD + row] = vv.e[e];
    }
    // hint next tiles toward the caches (global_prefetch_b8)
    if (kv0 + 64 < 1024) {
      int idx = tid * 8;
      int row = idx >> 6, col = idx & 63;
      __builtin_prefetch(&Kb[headBase + (size_t)(kv0 + 64 + row) * 64 + col], 0, 0);
      __builtin_prefetch(&Vb[headBase + (size_t)(kv0 + 64 + row) * 64 + col], 0, 0);
    }
    __syncthreads();

    // S = Q K^T  (16 q-rows x 64 kv-cols per wave)
    v8f Sf[4];
    #pragma unroll
    for (int j = 0; j < 4; ++j)
      for (int r = 0; r < 8; ++r) Sf[j][r] = 0.0f;
    #pragma unroll
    for (int ks = 0; ks < 64; ks += 32) {
      v16bf a = load_fragA(&Qs[qw * ATT_LD + ks], ATT_LD);
      #pragma unroll
      for (int j = 0; j < 4; ++j) {
        v16bf bfr = load_fragB(&Ks[(j * 16) * ATT_LD + ks], ATT_LD);
        Sf[j] = wmma_bf16(a, bfr, Sf[j]);
      }
    }

    // scale + additive mask log(clamp(cutoff, 1e-15))
    float sv[4][8];
    #pragma unroll
    for (int j = 0; j < 4; ++j)
      #pragma unroll
      for (int r = 0; r < 8; ++r) {
        int grow = q0 + qw + r + halfsel;
        int gcol = kv0 + j * 16 + col16;
        float c = cutoff[((size_t)b * 1024 + grow) * 1024 + gcol];
        sv[j][r] = Sf[j][r] * 0.125f + __logf(fmaxf(c, 1e-15f));
      }

    // online softmax: rows live in 16-lane halves (wave32)
    #pragma unroll
    for (int r = 0; r < 8; ++r) {
      float t = fmaxf(fmaxf(sv[0][r], sv[1][r]), fmaxf(sv[2][r], sv[3][r]));
      #pragma unroll
      for (int m = 1; m < 16; m <<= 1) t = fmaxf(t, __shfl_xor(t, m, 32));
      float mn    = fmaxf(mrow[r], t);
      float alpha = __expf(mrow[r] - mn);
      mrow[r] = mn;
      float ls = 0.0f;
      #pragma unroll
      for (int j = 0; j < 4; ++j) {
        float p = __expf(sv[j][r] - mn);
        ls += p;
        Ps[(qw + r + halfsel) * ATT_LD + j * 16 + col16] = f2bf(p);
      }
      #pragma unroll
      for (int m = 1; m < 16; m <<= 1) ls += __shfl_xor(ls, m, 32);
      lrow[r] = lrow[r] * alpha + ls;
      #pragma unroll
      for (int n = 0; n < 4; ++n) Of[n][r] *= alpha;
    }

    // O += P @ V   (Ps region is wave-private; LDS ops are in-order per wave)
    #pragma unroll
    for (int ks = 0; ks < 64; ks += 32) {
      v16bf a = load_fragA(&Ps[qw * ATT_LD + ks], ATT_LD);
      #pragma unroll
      for (int n = 0; n < 4; ++n) {
        v16bf bfr = load_fragB(&Vts[(n * 16) * ATT_LD + ks], ATT_LD);
        Of[n] = wmma_bf16(a, bfr, Of[n]);
      }
    }
  }

  // normalize and scatter to [B, S, D] bf16 (head-interleaved)
  #pragma unroll
  for (int r = 0; r < 8; ++r) {
    float inv = 1.0f / lrow[r];
    int grow = q0 + qw + r + halfsel;
    #pragma unroll
    for (int n = 0; n < 4; ++n) {
      size_t o = ((size_t)b * 1024 + grow) * 1024 + h * 64 + n * 16 + col16;
      attnb[o] = f2bf(Of[n][r] * inv);
    }
  }
}

// ---------------------------------------------------------------------------
// Kernel: output projection. out = attn @ Wo^T + bo   (f32 output)
// ---------------------------------------------------------------------------
__global__ void __launch_bounds__(256)
gemm_out(const bf16_t* __restrict__ attnb, const bf16_t* __restrict__ Wob,
         const float* __restrict__ bo, float* __restrict__ out) {
  __shared__ __align__(16) bf16_t As[2 * G_BUF];
  __shared__ __align__(16) bf16_t Bs[2 * G_BUF];
  const size_t n0 = (size_t)blockIdx.x * 128;
  const size_t m0 = (size_t)blockIdx.y * 128;
  v8f acc[4][2];
  gemm_core(attnb, Wob, 1024, 8192, 1024, m0, n0, As, Bs, acc);

  const int lane = threadIdx.x & 31;
  const int warp = threadIdx.x >> 5;
  const int wm = (warp >> 2) * 64, wn = (warp & 3) * 32;
  const int halfsel = (lane >> 4) << 3;
  const int col16   = lane & 15;
  #pragma unroll
  for (int mi = 0; mi < 4; ++mi)
    #pragma unroll
    for (int nj = 0; nj < 2; ++nj)
      #pragma unroll
      for (int r = 0; r < 8; ++r) {
        size_t gm = m0 + wm + mi * 16 + r + halfsel;
        size_t gn = n0 + wn + nj * 16 + col16;
        out[gm * 1024 + gn] = acc[mi][nj][r] + bo[gn];
      }
}

// ---------------------------------------------------------------------------
extern "C" void kernel_launch(void* const* d_in, const int* in_sizes, int n_in,
                              void* d_out, int out_size, void* d_ws, size_t ws_size,
                              hipStream_t stream) {
  const float* x      = (const float*)d_in[0];
  const float* cutoff = (const float*)d_in[1];
  const float* Wi     = (const float*)d_in[2];
  const float* bi     = (const float*)d_in[3];
  const float* Wo     = (const float*)d_in[4];
  const float* bo     = (const float*)d_in[5];
  float* out = (float*)d_out;

  const long long nx  = 8LL * 1024 * 1024;      // B*S*D
  const long long nwi = 3LL * 1024 * 1024;      // 3D*D
  const long long nwo = 1LL * 1024 * 1024;      // D*D

  bf16_t* p   = (bf16_t*)d_ws;
  bf16_t* xb  = p; p += nx;
  bf16_t* Wib = p; p += nwi;
  bf16_t* Wob = p; p += nwo;
  bf16_t* Qb  = p; p += nx;
  bf16_t* Kb  = p; p += nx;
  bf16_t* Vb  = p; p += nx;
  bf16_t* attnb = p;

  cvt_f32_bf16<<<(unsigned)(nx  / 1024), 256, 0, stream>>>(x,  xb,  nx);
  cvt_f32_bf16<<<(unsigned)(nwi / 1024), 256, 0, stream>>>(Wi, Wib, nwi);
  cvt_f32_bf16<<<(unsigned)(nwo / 1024), 256, 0, stream>>>(Wo, Wob, nwo);

  gemm_qkv<<<dim3(3072 / 128, 8192 / 128), 256, 0, stream>>>(xb, Wib, bi, Qb, Kb, Vb);
  attn_kernel<<<dim3(16, 16, 8), 128, 0, stream>>>(Qb, Kb, Vb, cutoff, attnb);
  gemm_out<<<dim3(1024 / 128, 8192 / 128), 256, 0, stream>>>(attnb, Wob, bo, out);
}
